// AdaptiveAttention_37082747634594
// MI455X (gfx1250) — matrix operand order repaired and checked by reading, not compiled, not run on hardware
//
#include <hip/hip_runtime.h>
#include <cstdint>
#include <cstddef>

#define B_   4
#define S_   1024
#define HID_ 1024
#define H_   16
#define D_   64

typedef __attribute__((ext_vector_type(16))) _Float16 v16h;
typedef __attribute__((ext_vector_type(8)))  _Float16 v8h;
typedef __attribute__((ext_vector_type(8)))  float    v8f;

union Frag16 { v16h v; v8h h[2]; };

// D = A(16x32 f16) * B(32x16 f16) + C(16x16 f32)
__device__ __forceinline__ v8f wmma32(const v16h& a, const v16h& b, const v8f& c) {
  return __builtin_amdgcn_wmma_f32_16x16x32_f16(false, a, false, b, (short)0, c,
                                                false, false);
}

// A-fragment (16x32, MxK) from row-major tile, row stride ld (halfs).
__device__ __forceinline__ v16h load_a16(const _Float16* tile, int ld, int lane) {
  const int m = lane & 15, g = (lane >> 4) & 1;
  const _Float16* p = tile + (size_t)m * ld + 16 * g;
  Frag16 f;
  f.h[0] = *(const v8h*)(p);
  f.h[1] = *(const v8h*)(p + 8);
  return f.v;
}

// B-fragment (32x16, KxN) from TRANSPOSED storage T[n][k], row stride ld.
__device__ __forceinline__ v16h load_bT16(const _Float16* tileT, int ld, int lane) {
  const int n = lane & 15, g = (lane >> 4) & 1;
  const _Float16* p = tileT + (size_t)n * ld + 16 * g;
  Frag16 f;
  f.h[0] = *(const v8h*)(p);
  f.h[1] = *(const v8h*)(p + 8);
  return f.v;
}

// ---- CDNA5 async global->LDS copy (ASYNCcnt), 16B per lane ----
__device__ __forceinline__ uint32_t lds_off_u32(const void* p) {
  return (uint32_t)(uintptr_t)p;  // low 32 bits of generic addr = LDS byte addr
}
__device__ __forceinline__ void async_cp16(_Float16* lds_dst,
                                           const _Float16* g_src) {
  asm volatile("global_load_async_to_lds_b128 %0, %1, off"
               :: "v"(lds_off_u32(lds_dst)), "v"(g_src)
               : "memory");
}
__device__ __forceinline__ void wait_async0() {
  asm volatile("s_wait_asynccnt 0" ::: "memory");
}

// ---------------- prep kernels ----------------
__global__ void cvt_f32_to_f16(const float* __restrict__ src,
                               _Float16* __restrict__ dst, int n) {
  int i = blockIdx.x * blockDim.x + threadIdx.x;
  if (i < n) dst[i] = (_Float16)src[i];
}

// WT[n*HID + k] = (f16) W[k*HID + n]
__global__ void transpose_to_f16(const float* __restrict__ W,
                                 _Float16* __restrict__ WT) {
  int i = blockIdx.x * blockDim.x + threadIdx.x;  // over HID*HID
  int nn = i >> 10;
  int kk = i & (HID_ - 1);
  WT[i] = (_Float16)W[(size_t)kk * HID_ + nn];
}

// ---------------- fused QKV projection ----------------
// One wave: 32 rows x 64 cols; 2-stage ping-pong pipeline (no reg rotation).
__global__ __launch_bounds__(128) void qkv_gemm(
    const _Float16* __restrict__ xh,
    const _Float16* __restrict__ WqT, const _Float16* __restrict__ WkT,
    const _Float16* __restrict__ WvT,
    const float* __restrict__ bq, const float* __restrict__ bk,
    const float* __restrict__ bv,
    _Float16* __restrict__ qb, _Float16* __restrict__ kb,
    _Float16* __restrict__ vt) {
  const int lane = threadIdx.x;
  const int wave = threadIdx.y;
  const int row0 = blockIdx.x * 32;
  const int col0 = (blockIdx.y * 4 + wave) * 64;
  const int mat  = blockIdx.z;
  const _Float16* WT = (mat == 0) ? WqT : (mat == 1) ? WkT : WvT;
  const float*  bias = (mat == 0) ? bq  : (mat == 1) ? bk  : bv;

  const _Float16* A0 = xh + (size_t)row0 * HID_;
  const _Float16* A1 = xh + (size_t)(row0 + 16) * HID_;

  v8f c[2][4] = {};
  // stage A @ k=0
  v16h a0A = load_a16(A0, HID_, lane);
  v16h a1A = load_a16(A1, HID_, lane);
  v16h bA[4];
#pragma unroll
  for (int t = 0; t < 4; ++t)
    bA[t] = load_bT16(WT + (size_t)(col0 + t * 16) * HID_, HID_, lane);

  for (int k0 = 0; k0 < HID_; k0 += 64) {
    // stage B loads @ k0+32
    v16h a0B = load_a16(A0 + k0 + 32, HID_, lane);
    v16h a1B = load_a16(A1 + k0 + 32, HID_, lane);
    v16h bB[4];
#pragma unroll
    for (int t = 0; t < 4; ++t)
      bB[t] = load_bT16(WT + (size_t)(col0 + t * 16) * HID_ + k0 + 32, HID_, lane);
    // compute stage A
#pragma unroll
    for (int t = 0; t < 4; ++t) {
      c[0][t] = wmma32(a0A, bA[t], c[0][t]);
      c[1][t] = wmma32(a1A, bA[t], c[1][t]);
    }
    // stage A loads @ k0+64 (clamped redundant prefetch on last iter)
    const int kn = (k0 + 64 < HID_) ? k0 + 64 : 0;
    a0A = load_a16(A0 + kn, HID_, lane);
    a1A = load_a16(A1 + kn, HID_, lane);
#pragma unroll
    for (int t = 0; t < 4; ++t)
      bA[t] = load_bT16(WT + (size_t)(col0 + t * 16) * HID_ + kn, HID_, lane);
    // compute stage B
#pragma unroll
    for (int t = 0; t < 4; ++t) {
      c[0][t] = wmma32(a0B, bB[t], c[0][t]);
      c[1][t] = wmma32(a1B, bB[t], c[1][t]);
    }
  }

  const int g = (lane >> 4) & 1, nn = lane & 15;
#pragma unroll
  for (int rg = 0; rg < 2; ++rg) {
#pragma unroll
    for (int t = 0; t < 4; ++t) {
      const int col = col0 + t * 16 + nn;
      const int h = col >> 6, d = col & 63;
      const float bcol = bias[col];
#pragma unroll
      for (int r = 0; r < 8; ++r) {
        const int row = row0 + rg * 16 + r + 8 * g;
        const int bb = row >> 10, ss = row & (S_ - 1);
        const _Float16 val = (_Float16)(c[rg][t][r] + bcol);
        if (mat == 0)
          qb[(((size_t)(bb * H_ + h)) * S_ + ss) * D_ + d] = val;
        else if (mat == 1)
          kb[(((size_t)(bb * H_ + h)) * S_ + ss) * D_ + d] = val;
        else
          vt[(((size_t)(bb * H_ + h)) * D_ + d) * S_ + ss] = val;
      }
    }
  }
}

// ---------------- flash attention ----------------
// Block = 4 waves sharing one (b,h), 64 queries.  K/V chunks (32 keys) are
// async-staged into double-buffered LDS once per block (4x traffic cut).
// Per wave: S^T = K.Q^T and O^T = V^T.P^T (one query per lane column).

// stage one 32-key chunk: K [32][64] (4KB contiguous) + V^T [64][32] (4KB)
__device__ __forceinline__ void stage_chunk(_Float16* kls, _Float16* vls,
                                            const _Float16* K,
                                            const _Float16* V,
                                            int key0, int tid) {
#pragma unroll
  for (int i = 0; i < 2; ++i) {
    const int l = tid + i * 128;  // 0..255 16B pieces each
    async_cp16(kls + l * 8, K + (size_t)key0 * D_ + l * 8);
    const int row = l >> 2, part = l & 3;
    async_cp16(vls + row * 32 + part * 8,
               V + (size_t)row * S_ + key0 + part * 8);
  }
}

__global__ __launch_bounds__(128) void flash_attn(
    const _Float16* __restrict__ qb, const _Float16* __restrict__ kb,
    const _Float16* __restrict__ vt, const unsigned char* __restrict__ mask,
    const float* __restrict__ temp, _Float16* __restrict__ ctx) {
  __shared__ __align__(16) _Float16 smem[2][4096];  // [buf][K 2048 | V 2048]
  const int lane = threadIdx.x;
  const int wave = threadIdx.y;
  const int tid  = wave * 32 + lane;
  const int blk  = blockIdx.x;               // B*H*(S/64)
  const int h    = (blk >> 4) & (H_ - 1);
  const int b    = blk >> 8;
  const int q0   = (blk & 15) * 64 + wave * 16;
  const int g = (lane >> 4) & 1, nn = lane & 15;

  const _Float16* Qt = qb + (((size_t)(b * H_ + h)) * S_ + q0) * D_;  // [q][d]
  const _Float16* K  = kb + ((size_t)(b * H_ + h)) * S_ * D_;         // [s][d]
  const _Float16* V  = vt + ((size_t)(b * H_ + h)) * D_ * S_;         // [d][s]
  const unsigned char* Mrow =
      mask + (((size_t)(b * H_ + h)) * S_ + (q0 + nn)) * S_;
  const float sc = temp[h] * 0.125f;  // temperature / sqrt(64)

  const v16h bq0 = load_bT16(Qt, D_, lane);       // Q^T as B operand
  const v16h bq1 = load_bT16(Qt + 32, D_, lane);

  v8f o[4] = {};            // O^T tiles: rows d = t*16 + r + 8g, col q = nn
  float m = -1e30f, l = 0.f;

  stage_chunk(&smem[0][0], &smem[0][2048], K, V, 0, tid);
  wait_async0();
  __syncthreads();

  int buf = 0;
  for (int key0 = 0; key0 < S_; key0 += 32) {
    const _Float16* kls = &smem[buf][0];
    const _Float16* vls = &smem[buf][2048];
    if (key0 + 32 < S_)   // block-uniform
      stage_chunk(&smem[buf ^ 1][0], &smem[buf ^ 1][2048], K, V, key0 + 32, tid);

    // scores^T from LDS: rows = keys (r+8g per tile), col = lane's query
    float sv[2][8];
#pragma unroll
    for (int nt = 0; nt < 2; ++nt) {
      v16h ka = load_a16(kls + (size_t)(nt * 16) * D_, D_, lane);
      v16h kc = load_a16(kls + (size_t)(nt * 16) * D_ + 32, D_, lane);
      v8f cs = {};
      cs = wmma32(ka, bq0, cs);
      cs = wmma32(kc, bq1, cs);
      const uint64_t mb = __builtin_nontemporal_load(
          (const uint64_t*)(Mrow + key0 + nt * 16 + 8 * g));
#pragma unroll
      for (int r = 0; r < 8; ++r) {
        const bool keep = ((mb >> (8 * r)) & 0xff) != 0;
        sv[nt][r] = keep ? cs[r] * sc : -1e30f;
      }
    }

    // online softmax: per-lane scalar stats, one cross-half exchange
    float mx = -1e30f;
#pragma unroll
    for (int nt = 0; nt < 2; ++nt)
#pragma unroll
      for (int r = 0; r < 8; ++r) mx = fmaxf(mx, sv[nt][r]);
    mx = fmaxf(mx, __shfl_xor(mx, 16, 32));
    const float mnew  = fmaxf(m, mx);
    const float alpha = __expf(m - mnew);
    float p[2][8];
    float rs = 0.f;
#pragma unroll
    for (int nt = 0; nt < 2; ++nt)
#pragma unroll
      for (int r = 0; r < 8; ++r) {
        p[nt][r] = __expf(sv[nt][r] - mnew);
        rs += p[nt][r];
      }
    rs += __shfl_xor(rs, 16, 32);
    l = l * alpha + rs;
    m = mnew;
#pragma unroll
    for (int t = 0; t < 4; ++t)
#pragma unroll
      for (int r = 0; r < 8; ++r) o[t][r] *= alpha;

    // Build P^T B-fragment (keys 16g..16g+15); 8 values come from the other
    // half-wave via shfl_xor(·,16) -- register-only transpose.
    float ex[8];
#pragma unroll
    for (int r = 0; r < 8; ++r) {
      const float send = g ? p[0][r] : p[1][r];
      ex[r] = __shfl_xor(send, 16, 32);
    }
    Frag16 pb;
#pragma unroll
    for (int e = 0; e < 8; ++e) {
      pb.v[e]     = (_Float16)(g ? ex[e] : p[0][e]);
      pb.v[e + 8] = (_Float16)(g ? p[1][e] : ex[e]);
    }

#pragma unroll
    for (int t = 0; t < 4; ++t) {
      v16h vf = load_a16(vls + (size_t)(t * 16) * 32, 32, lane);
      o[t] = wmma32(vf, pb.v, o[t]);
    }

    wait_async0();      // my staged copies for next chunk are in LDS
    __syncthreads();    // everyone done reading buf & staging buf^1
    buf ^= 1;
  }

  // normalize; lane holds one query row -> contiguous d, 16B vector stores
  const float inv = (l > 0.f) ? 1.f / l : 0.f;
  _Float16* crow = ctx + ((size_t)(b * S_ + q0 + nn)) * HID_ + h * D_;
#pragma unroll
  for (int t = 0; t < 4; ++t) {
    v8h st;
#pragma unroll
    for (int r = 0; r < 8; ++r) st[r] = (_Float16)(o[t][r] * inv);
    *(v8h*)(crow + t * 16 + 8 * g) = st;
  }
}

// ---------------- output projection ----------------
__global__ __launch_bounds__(128) void out_gemm(
    const _Float16* __restrict__ ctx, const _Float16* __restrict__ WoT,
    const float* __restrict__ bo, float* __restrict__ out) {
  const int lane = threadIdx.x;
  const int wave = threadIdx.y;
  const int row0 = blockIdx.x * 32;
  const int col0 = (blockIdx.y * 4 + wave) * 64;
  const _Float16* A0 = ctx + (size_t)row0 * HID_;
  const _Float16* A1 = ctx + (size_t)(row0 + 16) * HID_;

  v8f c[2][4] = {};
  v16h a0A = load_a16(A0, HID_, lane);
  v16h a1A = load_a16(A1, HID_, lane);
  v16h bA[4];
#pragma unroll
  for (int t = 0; t < 4; ++t)
    bA[t] = load_bT16(WoT + (size_t)(col0 + t * 16) * HID_, HID_, lane);

  for (int k0 = 0; k0 < HID_; k0 += 64) {
    v16h a0B = load_a16(A0 + k0 + 32, HID_, lane);
    v16h a1B = load_a16(A1 + k0 + 32, HID_, lane);
    v16h bB[4];
#pragma unroll
    for (int t = 0; t < 4; ++t)
      bB[t] = load_bT16(WoT + (size_t)(col0 + t * 16) * HID_ + k0 + 32, HID_, lane);
#pragma unroll
    for (int t = 0; t < 4; ++t) {
      c[0][t] = wmma32(a0A, bA[t], c[0][t]);
      c[1][t] = wmma32(a1A, bA[t], c[1][t]);
    }
    const int kn = (k0 + 64 < HID_) ? k0 + 64 : 0;
    a0A = load_a16(A0 + kn, HID_, lane);
    a1A = load_a16(A1 + kn, HID_, lane);
#pragma unroll
    for (int t = 0; t < 4; ++t)
      bA[t] = load_bT16(WoT + (size_t)(col0 + t * 16) * HID_ + kn, HID_, lane);
#pragma unroll
    for (int t = 0; t < 4; ++t) {
      c[0][t] = wmma32(a0B, bB[t], c[0][t]);
      c[1][t] = wmma32(a1B, bB[t], c[1][t]);
    }
  }

  const int g = (lane >> 4) & 1, nn = lane & 15;
#pragma unroll
  for (int rg = 0; rg < 2; ++rg)
#pragma unroll
    for (int t = 0; t < 4; ++t) {
      const int col = col0 + t * 16 + nn;
      const float bcol = bo[col];
#pragma unroll
      for (int r = 0; r < 8; ++r)
        out[((size_t)(row0 + rg * 16 + r + 8 * g)) * HID_ + col] =
            c[rg][t][r] + bcol;
    }
}

extern "C" void kernel_launch(void* const* d_in, const int* in_sizes, int n_in,
                              void* d_out, int out_size, void* d_ws, size_t ws_size,
                              hipStream_t stream) {
  const float* x    = (const float*)d_in[0];
  const float* Wq   = (const float*)d_in[1];
  const float* bq   = (const float*)d_in[2];
  const float* Wk   = (const float*)d_in[3];
  const float* bk   = (const float*)d_in[4];
  const float* Wv   = (const float*)d_in[5];
  const float* bv   = (const float*)d_in[6];
  const float* Wo   = (const float*)d_in[7];
  const float* bo   = (const float*)d_in[8];
  const float* temp = (const float*)d_in[9];
  const unsigned char* mask = (const unsigned char*)d_in[10];  // jax bool -> u8
  float* out = (float*)d_out;

  char* ws = (char*)d_ws;
  _Float16* xh  = (_Float16*)(ws);                         //  8 MiB
  _Float16* WqT = (_Float16*)(ws + ((size_t)8  << 20));    //  2 MiB
  _Float16* WkT = (_Float16*)(ws + ((size_t)10 << 20));
  _Float16* WvT = (_Float16*)(ws + ((size_t)12 << 20));
  _Float16* WoT = (_Float16*)(ws + ((size_t)14 << 20));
  _Float16* qb  = (_Float16*)(ws + ((size_t)16 << 20));    //  8 MiB [B,H,S,D]
  _Float16* kb  = (_Float16*)(ws + ((size_t)24 << 20));    //  8 MiB [B,H,S,D]
  _Float16* vt  = (_Float16*)(ws + ((size_t)32 << 20));    //  8 MiB [B,H,D,S]
  _Float16* ctx = (_Float16*)(ws + ((size_t)40 << 20));    //  8 MiB [B*S,HID]

  const int nX = B_ * S_ * HID_;
  cvt_f32_to_f16<<<(nX + 255) / 256, 256, 0, stream>>>(x, xh, nX);
  const int nW = HID_ * HID_;
  transpose_to_f16<<<nW / 256, 256, 0, stream>>>(Wq, WqT);
  transpose_to_f16<<<nW / 256, 256, 0, stream>>>(Wk, WkT);
  transpose_to_f16<<<nW / 256, 256, 0, stream>>>(Wv, WvT);
  transpose_to_f16<<<nW / 256, 256, 0, stream>>>(Wo, WoT);

  dim3 blk(32, 4);
  qkv_gemm<<<dim3(B_ * S_ / 32, HID_ / 256, 3), blk, 0, stream>>>(
      xh, WqT, WkT, WvT, bq, bk, bv, qb, kb, vt);
  flash_attn<<<dim3(B_ * H_ * (S_ / 64)), blk, 0, stream>>>(
      qb, kb, vt, mask, temp, ctx);
  out_gemm<<<dim3(B_ * S_ / 32, HID_ / 256), blk, 0, stream>>>(ctx, WoT, bo, out);
}
